// TimesformerCausalSelfAttention_35708358099555
// MI455X (gfx1250) — compile-verified, matrix operand order checked by reading, not run on hardware
//
#include <hip/hip_runtime.h>

// MI455X / gfx1250, wave32, WMMA bf16 -> f32 accumulate.
// B=32, S=1024, C=768, H=12, D=64, scale = 1/8.

typedef __attribute__((ext_vector_type(16))) __bf16 v16bf;
typedef __attribute__((ext_vector_type(8)))  __bf16 v8bf;
typedef __attribute__((ext_vector_type(8)))  float  v8f;
typedef __attribute__((ext_vector_type(4)))  int    v4i;

static __device__ __forceinline__ v16bf concat8(v8bf lo, v8bf hi) {
  return __builtin_shufflevector(lo, hi, 0,1,2,3,4,5,6,7,8,9,10,11,12,13,14,15);
}

static __device__ __forceinline__ v8f wmma_bf16(v16bf a, v16bf b, v8f c) {
  return __builtin_amdgcn_wmma_f32_16x16x32_bf16(false, a, false, b, (short)0, c, false, false);
}

// Async global->LDS 16B copy (CDNA5 GLOBAL_LOAD_ASYNC_TO_LDS_B128, ASYNCcnt).
// Probe: builtin exists with signature (global v4i*, local v4i*, imm, imm).
static __device__ __forceinline__ void async_copy16(const __bf16* g, __bf16* l) {
#if defined(__HIP_DEVICE_COMPILE__)
#if __has_builtin(__builtin_amdgcn_global_load_async_to_lds_b128)
  __builtin_amdgcn_global_load_async_to_lds_b128(
      (__attribute__((address_space(1))) v4i*)g,
      (__attribute__((address_space(3))) v4i*)l, 0, 0);
#else
  asm volatile("global_load_async_to_lds_b128 %0, %1, off"
               :
               : "v"((__attribute__((address_space(3))) v4i*)l), "v"(g)
               : "memory");
#endif
#endif
}

static __device__ __forceinline__ void wait_async0() {
#if defined(__HIP_DEVICE_COMPILE__)
#if __has_builtin(__builtin_amdgcn_s_wait_asynccnt)
  __builtin_amdgcn_s_wait_asynccnt(0);
#else
  asm volatile("s_wait_asynccnt 0x0" ::: "memory");
#endif
#endif
}

// ---------------------------------------------------------------------------
// Kernel 1: f32 -> bf16 conversion (vectorized x4)
// ---------------------------------------------------------------------------
__global__ __launch_bounds__(256) void cvt_f32_bf16(const float* __restrict__ src,
                                                    __bf16* __restrict__ dst, int n) {
  int i = (blockIdx.x * 256 + threadIdx.x) * 4;
  if (i + 4 <= n) {
    float4 f = *reinterpret_cast<const float4*>(src + i);
    __bf16 o[4] = {(__bf16)f.x, (__bf16)f.y, (__bf16)f.z, (__bf16)f.w};
    *reinterpret_cast<uint2*>(dst + i) = *reinterpret_cast<const uint2*>(o);
  }
}

// ---------------------------------------------------------------------------
// Kernel 2: QKV projection GEMM.  out[m,n] = sum_k X[m,k] * W[n,k] + b[n]
// M=32768, N=2304, K=768.  Block tile 128x128, 8 waves, wave tile 32x64.
// Async-to-LDS double-buffered staging; epilogue routes n ->
// Q[b,h,s,d] / K[b,h,s,d] / Vt[b,h,d,s] in bf16.
// ---------------------------------------------------------------------------
#define LDT 56   // LDS row stride in bf16 elems (112 B: 16B-aligned, conflict-free)

__global__ __launch_bounds__(256) void qkv_gemm(
    const __bf16* __restrict__ X,   // [32768,768]
    const __bf16* __restrict__ W,   // [2304,768]
    const float*  __restrict__ bias,// [2304]
    __bf16* __restrict__ Q,         // [B,H,S,D]
    __bf16* __restrict__ K,         // [B,H,S,D]
    __bf16* __restrict__ Vt)        // [B,H,D,S]
{
  __shared__ __bf16 lA[2][128 * LDT];
  __shared__ __bf16 lB[2][128 * LDT];

  const int m0   = blockIdx.x * 128;
  const int n0   = blockIdx.y * 128;
  const int tid  = threadIdx.x;
  const int lane = tid & 31;
  const int wave = tid >> 5;
  const int half = lane >> 4;      // 0: lanes 0-15, 1: lanes 16-31
  const int l16  = lane & 15;
  const int wm   = (wave & 3) * 32;  // 4 waves along M
  const int wn   = (wave >> 2) * 64; // 2 waves along N

  // 1024 16B chunks per stage (A: 512, B: 512); 4 async copies per thread.
  auto issueTiles = [&](int buf, int k0) {
#pragma unroll
    for (int u = 0; u < 2; ++u) {
      const int c   = tid + u * 256;
      const int row = c >> 2, cb = c & 3;
      async_copy16(X + (size_t)(m0 + row) * 768 + k0 + cb * 8,
                   &lA[buf][row * LDT + cb * 8]);
      async_copy16(W + (size_t)(n0 + row) * 768 + k0 + cb * 8,
                   &lB[buf][row * LDT + cb * 8]);
    }
  };

  v8f acc[2][4] = {};

  issueTiles(0, 0);

  for (int kt = 0; kt < 24; ++kt) {
    const int buf = kt & 1;
    wait_async0();        // this wave's async tile copies are in LDS
    __syncthreads();      // publish across the workgroup
    if (kt + 1 < 24) issueTiles(buf ^ 1, (kt + 1) * 32);  // overlap next stage

    v16bf af[2], bfm[4];
#pragma unroll
    for (int i = 0; i < 2; ++i) {
      const __bf16* base = &lA[buf][(wm + i * 16 + l16) * LDT] + half * 8;
      af[i] = concat8(*reinterpret_cast<const v8bf*>(base),
                      *reinterpret_cast<const v8bf*>(base + 16));
    }
#pragma unroll
    for (int j = 0; j < 4; ++j) {
      const __bf16* base = &lB[buf][(wn + j * 16 + l16) * LDT] + half * 8;
      bfm[j] = concat8(*reinterpret_cast<const v8bf*>(base),
                       *reinterpret_cast<const v8bf*>(base + 16));
    }
#pragma unroll
    for (int i = 0; i < 2; ++i)
#pragma unroll
      for (int j = 0; j < 4; ++j)
        acc[i][j] = wmma_bf16(af[i], bfm[j], acc[i][j]);
  }

  // Epilogue: + bias, convert bf16, scatter into Q / K / Vt layouts.
  float bn[4];
#pragma unroll
  for (int j = 0; j < 4; ++j) bn[j] = bias[n0 + wn + j * 16 + l16];

#pragma unroll
  for (int i = 0; i < 2; ++i) {
#pragma unroll
    for (int j = 0; j < 4; ++j) {
      const int n  = n0 + wn + j * 16 + l16;
      const int t  = n / 768;
      const int nn = n % 768;
      const int h  = nn >> 6;
      const int dd = nn & 63;
#pragma unroll
      for (int r = 0; r < 8; ++r) {
        const int m = m0 + wm + i * 16 + r + 8 * half;
        const int b = m >> 10;
        const int s = m & 1023;
        const size_t bh = (size_t)b * 12 + h;
        const __bf16 val = (__bf16)(acc[i][j][r] + bn[j]);
        if (t == 0)      Q[(bh * 1024 + s) * 64 + dd] = val;
        else if (t == 1) K[(bh * 1024 + s) * 64 + dd] = val;
        else             Vt[(bh * 64 + dd) * 1024 + s] = val;
      }
    }
  }
}

// ---------------------------------------------------------------------------
// Kernel 3: fused causal attention (flash-style, online softmax).
// Grid: (S/64, B*H), 128 threads (4 waves); each wave owns a 16-row q tile.
// KV step = 32 keys: scores 16x32 (2 wmma n-tiles), P@V with one k=32 wmma
// per 16-wide d tile (4 tiles for d=64).  P transposed C->A layout via LDS.
// ---------------------------------------------------------------------------
__global__ __launch_bounds__(128) void attn_causal(
    const __bf16* __restrict__ Q,  // [B,H,S,D]
    const __bf16* __restrict__ K,  // [B,H,S,D]
    const __bf16* __restrict__ Vt, // [B,H,D,S]
    float* __restrict__ out)       // [B,S,C]
{
  __shared__ __bf16 lP[4][16 * LDT];  // per-wave 16 x 32 bf16, padded stride

  const int tid  = threadIdx.x;
  const int lane = tid & 31;
  const int wave = tid >> 5;
  const int half = lane >> 4;
  const int l16  = lane & 15;

  const int bh = blockIdx.y;               // 0..383
  const int qt = blockIdx.x * 4 + wave;    // q tile 0..63
  const int qs = qt * 16;

  const __bf16* Qb = Q  + (size_t)bh * 1024 * 64;
  const __bf16* Kb = K  + (size_t)bh * 1024 * 64;
  const __bf16* Vb = Vt + (size_t)bh * 64 * 1024;

  // Q fragments (rows qs..qs+15, d split in two k=32 steps), loaded once.
  v16bf qa[2];
#pragma unroll
  for (int ks = 0; ks < 2; ++ks) {
    const __bf16* base = Qb + (size_t)(qs + l16) * 64 + ks * 32 + half * 8;
    qa[ks] = concat8(*reinterpret_cast<const v8bf*>(base),
                     *reinterpret_cast<const v8bf*>(base + 16));
  }

  float mrow[8], lrow[8];
  v8f ctx[4] = {};
#pragma unroll
  for (int r = 0; r < 8; ++r) { mrow[r] = -1e30f; lrow[r] = 0.0f; }

  const int ntiles = (qs + 16 + 31) >> 5;  // causal limit in 32-key tiles

  for (int kv = 0; kv < ntiles; ++kv) {
    const int k0 = kv * 32;

    // Prefetch next KV tile into cache (global_prefetch_b8).
    if (kv + 1 < ntiles) {
      __builtin_prefetch(Kb + (size_t)(k0 + 32 + lane) * 64, 0, 0);
      __builtin_prefetch(Vb + (size_t)lane * 1024 + k0 + 32, 0, 0);
      __builtin_prefetch(Vb + (size_t)(lane + 32) * 1024 + k0 + 32, 0, 0);
    }

    // ---- S = (Q @ K^T) * scale, causal mask ----
    v8f s0 = {}, s1 = {};
#pragma unroll
    for (int ks = 0; ks < 2; ++ks) {
      const __bf16* b0 = Kb + (size_t)(k0 + l16) * 64 + ks * 32 + half * 8;
      v16bf kb0 = concat8(*reinterpret_cast<const v8bf*>(b0),
                          *reinterpret_cast<const v8bf*>(b0 + 16));
      s0 = wmma_bf16(qa[ks], kb0, s0);
      const __bf16* b1 = Kb + (size_t)(k0 + 16 + l16) * 64 + ks * 32 + half * 8;
      v16bf kb1 = concat8(*reinterpret_cast<const v8bf*>(b1),
                          *reinterpret_cast<const v8bf*>(b1 + 16));
      s1 = wmma_bf16(qa[ks], kb1, s1);
    }

    const int qg  = qs + 8 * half;      // + r below
    const int kg0 = k0 + l16;
    const int kg1 = k0 + 16 + l16;
#pragma unroll
    for (int r = 0; r < 8; ++r) {
      s0[r] = (kg0 <= qg + r) ? s0[r] * 0.125f : -1e30f;
      s1[r] = (kg1 <= qg + r) ? s1[r] * 0.125f : -1e30f;
    }

    // ---- online softmax (row stats via 16-lane xor shuffles) ----
#pragma unroll
    for (int r = 0; r < 8; ++r) {
      float mx = fmaxf(s0[r], s1[r]);
#pragma unroll
      for (int off = 8; off >= 1; off >>= 1) mx = fmaxf(mx, __shfl_xor(mx, off, 32));
      const float mn = fmaxf(mrow[r], mx);
      const float alpha = __expf(mrow[r] - mn);
      mrow[r] = mn;
      s0[r] = __expf(s0[r] - mn);
      s1[r] = __expf(s1[r] - mn);
      float rs = s0[r] + s1[r];
#pragma unroll
      for (int off = 8; off >= 1; off >>= 1) rs += __shfl_xor(rs, off, 32);
      lrow[r] = lrow[r] * alpha + rs;
#pragma unroll
      for (int t = 0; t < 4; ++t) ctx[t][r] *= alpha;
    }

    // ---- P: C-layout -> A-layout transpose through per-wave LDS ----
    __bf16* lp = &lP[wave][0];
#pragma unroll
    for (int r = 0; r < 8; ++r) {
      const int row = r + 8 * half;
      lp[row * LDT + l16]      = (__bf16)s0[r];
      lp[row * LDT + 16 + l16] = (__bf16)s1[r];
    }
    __builtin_amdgcn_wave_barrier();
    asm volatile("s_wait_dscnt 0x0" ::: "memory");
    __builtin_amdgcn_wave_barrier();

    const __bf16* pbase = lp + l16 * LDT + half * 8;
    v16bf pa = concat8(*reinterpret_cast<const v8bf*>(pbase),
                       *reinterpret_cast<const v8bf*>(pbase + 16));

    // ---- ctx += P @ V  (V^T rows are key-contiguous) ----
#pragma unroll
    for (int t = 0; t < 4; ++t) {
      const __bf16* vbase = Vb + (size_t)(t * 16 + l16) * 1024 + k0 + half * 8;
      v16bf vb = concat8(*reinterpret_cast<const v8bf*>(vbase),
                         *reinterpret_cast<const v8bf*>(vbase + 16));
      ctx[t] = wmma_bf16(pa, vb, ctx[t]);
    }
  }

  // ---- normalize and write f32 output [B,S,C] ----
  const int b = bh / 12;
  const int h = bh % 12;
#pragma unroll
  for (int t = 0; t < 4; ++t) {
#pragma unroll
    for (int r = 0; r < 8; ++r) {
      const int qrow = qs + r + 8 * half;
      const int cc   = h * 64 + t * 16 + l16;
      out[((size_t)b * 1024 + qrow) * 768 + cc] = ctx[t][r] / lrow[r];
    }
  }
}

// ---------------------------------------------------------------------------
// Launch
// ---------------------------------------------------------------------------
extern "C" void kernel_launch(void* const* d_in, const int* in_sizes, int n_in,
                              void* d_out, int out_size, void* d_ws, size_t ws_size,
                              hipStream_t stream) {
  const float* hs    = (const float*)d_in[0];  // [32,1024,768]
  const float* qkv_w = (const float*)d_in[1];  // [2304,768]
  const float* qkv_b = (const float*)d_in[2];  // [2304]
  float* out = (float*)d_out;

  char* ws = (char*)d_ws;
  __bf16* Xbf = (__bf16*)(ws);                          // 50331648 B
  __bf16* Wbf = (__bf16*)(ws + 50331648);               //  3538944 B
  __bf16* Qbf = (__bf16*)(ws + 53870592);               // 50331648 B
  __bf16* Kbf = (__bf16*)(ws + 104202240);              // 50331648 B
  __bf16* Vtb = (__bf16*)(ws + 154533888);              // 50331648 B

  const int nX = 32 * 1024 * 768;   // 25165824
  const int nW = 2304 * 768;        //  1769472
  cvt_f32_bf16<<<nX / 1024, 256, 0, stream>>>(hs, Xbf, nX);
  cvt_f32_bf16<<<nW / 1024, 256, 0, stream>>>(qkv_w, Wbf, nW);

  qkv_gemm<<<dim3(256, 18), 256, 0, stream>>>(Xbf, Wbf, qkv_b, Qbf, Kbf, Vtb);

  attn_causal<<<dim3(16, 384), 128, 0, stream>>>(Qbf, Kbf, Vtb, out);
}
